// S4DLayer_21543555957240
// MI455X (gfx1250) — matrix operand order repaired and checked by reading, not compiled, not run on hardware
//
#include <hip/hip_runtime.h>
#include <hip/hip_bf16.h>

// ---------------------------------------------------------------------------
// S4D layer, chunked-scan formulation for MI455X (gfx1250, wave32, WMMA).
//   B=8, H=128, N=64, L=8192.  Chunk T=64, NCH=128, J=B*NCH=1024 cols/head.
//   y = Re(Er @ S) + Toe @ x_chunk + D*x,  S scanned with decay exp(T*dA).
// Matmuls on v_wmma_f32_16x16x32_bf16 (f32 accumulate). One wave owns a
// 16-column tile and sweeps all M-tiles so B fragments are loaded once.
// ---------------------------------------------------------------------------

#define Bc   8
#define Hc   128
#define Nc_  64      // SSM state size N
#define Lc   8192
#define Tc   64      // chunk length
#define NCH  128     // number of chunks = L/T
#define Jc   1024    // B * NCH columns per head
#define NN   128     // 2N (complex packed as real rows)

typedef __attribute__((ext_vector_type(16))) __bf16 v16bf;
typedef __attribute__((ext_vector_type(8)))  float  v8f;

// workspace layout (bytes)
static constexpr size_t OFF_G   = 0;                              // bf16 H*NN*T
static constexpr size_t OFF_ER  = OFF_G   + 2ull*Hc*NN*Tc;        // bf16 H*T*NN
static constexpr size_t OFF_TOE = OFF_ER  + 2ull*Hc*Tc*NN;        // bf16 H*T*T
static constexpr size_t OFF_DEC = OFF_TOE + 2ull*Hc*Tc*Tc;        // f32  H*N*2
static constexpr size_t OFF_X   = OFF_DEC + 4ull*Hc*Nc_*2;        // bf16 H*J*T
static constexpr size_t OFF_U   = OFF_X   + 2ull*Hc*Jc*Tc;        // f32  H*J*NN
static constexpr size_t OFF_S   = OFF_U   + 4ull*Hc*Jc*NN;        // bf16 H*J*NN

// ---- WMMA fragment helpers (contiguous k = 16*g + q order, same for A and B) --------------
__device__ inline v8f wmma_bf16(v16bf a, v16bf b, v8f c) {
  return __builtin_amdgcn_wmma_f32_16x16x32_bf16(
      false, a, false, b, (short)0, c, false, false);
}

// A-matrix 16x32: lane m = lane&15; halves q -> K = g*16+q; g = lane>>4.
// Row-major src [m][k].
__device__ inline v16bf load_a_frag(const __bf16* __restrict__ base, int ld, int lane) {
  const int g = lane >> 4, m = lane & 15;
  v16bf a;
#pragma unroll
  for (int q = 0; q < 16; ++q) a[q] = base[(size_t)m * ld + (g * 16 + q)];
  return a;
}

// B-matrix 32x16: lane n = lane&15; halves q -> K = g*16+q. Src is
// column-major [n][k] (k contiguous), so each lane reads 32 contiguous bytes.
__device__ inline v16bf load_b_frag(const __bf16* __restrict__ base, int ld, int lane) {
  const int g = lane >> 4, n = lane & 15;
  v16bf b;
#pragma unroll
  for (int q = 0; q < 16; ++q) b[q] = base[(size_t)n * ld + (g * 16 + q)];
  return b;
}

// ---- K1: per-head kernel generation (dA, W, G, Er, Toeplitz, decay) -------
__global__ void __launch_bounds__(64)
s4d_gen_kernel(const float* __restrict__ A_re, const float* __restrict__ A_im,
               const float* __restrict__ C,    const float* __restrict__ log_delta,
               __bf16* __restrict__ G, __bf16* __restrict__ Er,
               __bf16* __restrict__ Toe, float* __restrict__ dec) {
  __shared__ float s_dre[Nc_], s_dim[Nc_], s_wre[Nc_], s_wim[Nc_], s_k[Tc];
  const int h = blockIdx.x, t = threadIdx.x;          // t doubles as n, tau, u

  // per-state constants
  const float are  = fminf(A_re[h * Nc_ + t], -1e-4f);
  const float aim  = A_im[h * Nc_ + t];
  const float step = __expf(log_delta[h]);
  const float dre = step * are, dim = step * aim;
  s_dre[t] = dre; s_dim[t] = dim;
  // W = Cc * (exp(dA)-1)/A
  float sn, cs;  __sincosf(dim, &sn, &cs);
  const float em  = __expf(dre);
  const float nre = em * cs - 1.0f, nim = em * sn;
  const float den = are * are + aim * aim;
  const float qre = (nre * are + nim * aim) / den;
  const float qim = (nim * are - nre * aim) / den;
  const float c0 = C[(h * Nc_ + t) * 2 + 0], c1 = C[(h * Nc_ + t) * 2 + 1];
  s_wre[t] = c0 * qre - c1 * qim;
  s_wim[t] = c0 * qim + c1 * qre;
  // decay = exp(T * dA)
  { float s2, c2; __sincosf((float)Tc * dim, &s2, &c2);
    const float m2 = __expf((float)Tc * dre);
    dec[(h * Nc_ + t) * 2 + 0] = m2 * c2;
    dec[(h * Nc_ + t) * 2 + 1] = m2 * s2; }
  __syncthreads();

  float kacc = 0.0f;
#pragma unroll 4
  for (int n = 0; n < Nc_; ++n) {
    const float dr = s_dre[n], di = s_dim[n], wr = s_wre[n], wi = s_wim[n];
    // K[t] = Re(w * exp(t*dA))
    { float s, c; __sincosf((float)t * di, &s, &c);
      kacc += __expf((float)t * dr) * (wr * c - wi * s); }
    // Er[t][n] = Re(w*exp((t+1)dA)),  Er[t][N+n] = -Im(...)
    { float s, c; __sincosf((float)(t + 1) * di, &s, &c);
      const float m = __expf((float)(t + 1) * dr);
      Er[((size_t)h * Tc + t) * NN + n]        = (__bf16)( m * (wr * c - wi * s));
      Er[((size_t)h * Tc + t) * NN + Nc_ + n]  = (__bf16)(-m * (wr * s + wi * c)); }
    // G[n][u] = Re(exp((T-1-u)dA)),  G[N+n][u] = Im(...)
    { float s, c; __sincosf((float)(Tc - 1 - t) * di, &s, &c);
      const float m = __expf((float)(Tc - 1 - t) * dr);
      G[((size_t)h * NN + n) * Tc + t]        = (__bf16)(m * c);
      G[((size_t)h * NN + Nc_ + n) * Tc + t]  = (__bf16)(m * s); }
  }
  s_k[t] = kacc;
  __syncthreads();
  // Toeplitz row t: Toe[t][c] = (c<=t) ? K[t-c] : 0
#pragma unroll 4
  for (int c = 0; c < Tc; ++c)
    Toe[((size_t)h * Tc + t) * Tc + c] = (__bf16)((c <= t) ? s_k[t - c] : 0.0f);
}

// ---- K2: pack x -> bf16, layout [h][j=(b*NCH+c)][tau] ---------------------
__global__ void __launch_bounds__(256)
s4d_pack_x(const float* __restrict__ x, __bf16* __restrict__ X) {
  const size_t idx = (size_t)blockIdx.x * 256 + threadIdx.x;
  if (idx >= (size_t)Bc * Hc * Lc) return;
  const int l = (int)(idx % Lc);
  const int h = (int)((idx / Lc) % Hc);
  const int b = (int)(idx / ((size_t)Lc * Hc));
  const int c = l >> 6, tau = l & 63;
  X[((size_t)h * Jc + (b * NCH + c)) * Tc + tau] = (__bf16)x[idx];
}

// ---- K3: chunk-state matmul  U(2N x J) = G(2N x T) @ X(T x J) per head ----
// One wave owns a 16-column tile of X (B frags live in VGPRs), sweeps all
// eight 16-row tiles of G.  X read exactly once.
__global__ void __launch_bounds__(128)
s4d_state_mm(const __bf16* __restrict__ G, const __bf16* __restrict__ X,
             float* __restrict__ U) {
  const int lane = threadIdx.x & 31;
  const int tid  = blockIdx.x * 4 + (threadIdx.x >> 5); // H*64 wave tiles
  const int h  = tid >> 6;
  const int jt = tid & 63;
  const __bf16* Bb = X + ((size_t)h * Jc + jt * 16) * Tc;
  const v16bf b0 = load_b_frag(Bb +  0, Tc, lane);
  const v16bf b1 = load_b_frag(Bb + 32, Tc, lane);
  const int g = lane >> 4, col = lane & 15;
  float* Ucol = U + ((size_t)h * Jc + jt * 16 + col) * NN;
#pragma unroll
  for (int mt = 0; mt < 8; ++mt) {
    const __bf16* Ab = G + ((size_t)h * NN + mt * 16) * Tc;
    if (mt < 7) __builtin_prefetch(Ab + 16 * Tc, 0, 1);   // global_prefetch_b8
    v8f acc = {};
    acc = wmma_bf16(load_a_frag(Ab +  0, Tc, lane), b0, acc);
    acc = wmma_bf16(load_a_frag(Ab + 32, Tc, lane), b1, acc);
#pragma unroll
    for (int r = 0; r < 8; ++r) Ucol[mt * 16 + 8 * g + r] = acc[r];
  }
}

// ---- K4: cross-chunk scan  S_{c+1} = decay*S_c + U_c ; store bf16 states --
__global__ void __launch_bounds__(256)
s4d_scan(const float* __restrict__ U, const float* __restrict__ dec,
         __bf16* __restrict__ S) {
  const int t = blockIdx.x * 256 + threadIdx.x;     // H*B*N = 65536 threads
  const int h = t >> 9, b = (t >> 6) & 7, n = t & 63;
  const float dre = dec[(h * Nc_ + n) * 2 + 0];
  const float dim = dec[(h * Nc_ + n) * 2 + 1];
  float sre = 0.0f, sim = 0.0f;
  for (int c = 0; c < NCH; ++c) {
    const size_t base = ((size_t)h * Jc + (b * NCH + c)) * NN;
    S[base + n]       = (__bf16)sre;               // state BEFORE chunk c
    S[base + Nc_ + n] = (__bf16)sim;
    const float ure = U[base + n], uim = U[base + Nc_ + n];
    const float nre = dre * sre - dim * sim + ure;
    sim = dre * sim + dim * sre + uim;
    sre = nre;
  }
}

// ---- K5: outputs  Y = Er @ S + Toe @ X ; out = Y + D*x --------------------
// One wave owns a 16-column tile: S frags (4) + X frags (2) loaded once into
// VGPRs, then sweeps the four 16-row tiles of Er/Toe.
__global__ void __launch_bounds__(128)
s4d_out_mm(const __bf16* __restrict__ Er, const __bf16* __restrict__ S,
           const __bf16* __restrict__ Toe, const __bf16* __restrict__ X,
           const float* __restrict__ x, const float* __restrict__ D,
           float* __restrict__ out) {
  const int lane = threadIdx.x & 31;
  const int tid  = blockIdx.x * 4 + (threadIdx.x >> 5); // H*64 wave tiles
  const int h  = tid >> 6;
  const int jt = tid & 63;
  const __bf16* BbS = S + ((size_t)h * Jc + jt * 16) * NN;
  const __bf16* BbX = X + ((size_t)h * Jc + jt * 16) * Tc;
  const v16bf bs0 = load_b_frag(BbS +  0, NN, lane);
  const v16bf bs1 = load_b_frag(BbS + 32, NN, lane);
  const v16bf bs2 = load_b_frag(BbS + 64, NN, lane);
  const v16bf bs3 = load_b_frag(BbS + 96, NN, lane);
  const v16bf bx0 = load_b_frag(BbX +  0, Tc, lane);
  const v16bf bx1 = load_b_frag(BbX + 32, Tc, lane);

  const int g = lane >> 4, col = lane & 15;
  const int j = jt * 16 + col, b = j >> 7, c = j & 127;
  const float Dh = D[h];
  const float* xcol = x + ((size_t)b * Hc + h) * Lc + (size_t)c * Tc;
  float* ocol = out + ((size_t)b * Hc + h) * Lc + (size_t)c * Tc;

#pragma unroll
  for (int mt = 0; mt < 4; ++mt) {
    const __bf16* AbE = Er  + ((size_t)h * Tc + mt * 16) * NN;
    const __bf16* AbT = Toe + ((size_t)h * Tc + mt * 16) * Tc;
    if (mt < 3) { __builtin_prefetch(AbE + 16 * NN, 0, 1);
                  __builtin_prefetch(AbT + 16 * Tc, 0, 1); }
    v8f acc = {};
    acc = wmma_bf16(load_a_frag(AbE +  0, NN, lane), bs0, acc);
    acc = wmma_bf16(load_a_frag(AbE + 32, NN, lane), bs1, acc);
    acc = wmma_bf16(load_a_frag(AbE + 64, NN, lane), bs2, acc);
    acc = wmma_bf16(load_a_frag(AbE + 96, NN, lane), bs3, acc);
    acc = wmma_bf16(load_a_frag(AbT +  0, Tc, lane), bx0, acc);
    acc = wmma_bf16(load_a_frag(AbT + 32, Tc, lane), bx1, acc);
#pragma unroll
    for (int r = 0; r < 8; ++r) {
      const int row = mt * 16 + 8 * g + r;
      ocol[row] = acc[r] + Dh * xcol[row];
    }
  }
}

// ---------------------------------------------------------------------------
extern "C" void kernel_launch(void* const* d_in, const int* in_sizes, int n_in,
                              void* d_out, int out_size, void* d_ws, size_t ws_size,
                              hipStream_t stream) {
  const float* x    = (const float*)d_in[0];
  const float* A_re = (const float*)d_in[1];
  const float* A_im = (const float*)d_in[2];
  const float* C    = (const float*)d_in[3];
  const float* D    = (const float*)d_in[4];
  const float* ld_  = (const float*)d_in[5];
  float* out = (float*)d_out;

  char* ws = (char*)d_ws;
  __bf16* G   = (__bf16*)(ws + OFF_G);
  __bf16* Er  = (__bf16*)(ws + OFF_ER);
  __bf16* Toe = (__bf16*)(ws + OFF_TOE);
  float*  dec = (float* )(ws + OFF_DEC);
  __bf16* X   = (__bf16*)(ws + OFF_X);
  float*  U   = (float* )(ws + OFF_U);
  __bf16* S   = (__bf16*)(ws + OFF_S);

  s4d_gen_kernel<<<Hc, 64, 0, stream>>>(A_re, A_im, C, ld_, G, Er, Toe, dec);
  s4d_pack_x<<<((size_t)Bc * Hc * Lc + 255) / 256, 256, 0, stream>>>(x, X);
  s4d_state_mm<<<(Hc * 64) / 4, 128, 0, stream>>>(G, X, U);         // 2048 blk
  s4d_scan<<<(Hc * Bc * Nc_) / 256, 256, 0, stream>>>(U, dec, S);   //  256 blk
  s4d_out_mm<<<(Hc * 64) / 4, 128, 0, stream>>>(Er, S, Toe, X, x, D, out);
}
